// Seatt_semantic_selector_876173328810
// MI455X (gfx1250) — matrix operand order repaired and checked by reading, not compiled, not run on hardware
//
#include <hip/hip_runtime.h>
#include <stdint.h>

typedef __attribute__((ext_vector_type(16))) __bf16 v16bf;
typedef __attribute__((ext_vector_type(8)))  __bf16 v8bf;
typedef __attribute__((ext_vector_type(8)))  float  v8f;

#define ATT_SCALE 0.17677669529663687f  // 32^-0.5

// CDNA5 async global->LDS copy path (gfx1250). Guarded so compile never breaks.
#if defined(__HIP_DEVICE_COMPILE__) && \
    __has_builtin(__builtin_amdgcn_global_load_async_to_lds_b128) && \
    __has_builtin(__builtin_amdgcn_s_wait_asynccnt)
#define HAVE_ASYNC_LDS 1
#else
#define HAVE_ASYNC_LDS 0
#endif

#if HAVE_ASYNC_LDS
// exact parameter types per hipcc diagnostic: int __vector(4) pointers,
// global (AS1) source and LDS (AS3) destination
typedef int ivec4 __attribute__((vector_size(16)));
typedef __attribute__((address_space(1))) ivec4 g_ivec4;
typedef __attribute__((address_space(3))) ivec4 l_ivec4;

static __device__ __forceinline__ void async_copy_b128(const void* g, void* l) {
  __builtin_amdgcn_global_load_async_to_lds_b128(
      (g_ivec4*)(void*)g, (l_ivec4*)l, 0, 0);
}
#endif

// ---------------------------------------------------------------------------
// WMMA fragment loaders (wave32; k = 16*(lane/16) + e in A and B alike)
// ---------------------------------------------------------------------------
static __device__ __forceinline__ v16bf frag_a_load(const __bf16* a, int ld) {
  const int lane = threadIdx.x & 31;
  const int m  = lane & 15;
  const int kh = lane >> 4;
  v16bf f;
#pragma unroll
  for (int e = 0; e < 16; ++e) {
    const int kk = kh * 16 + e;
    f[e] = a[m * ld + kk];
  }
  return f;
}

// B 32x16 bf16 from an LDS tile stored [n][k] (row = output column)
static __device__ __forceinline__ v16bf frag_b_load_nk(const __bf16* b, int ld) {
  const int lane = threadIdx.x & 31;
  const int n  = lane & 15;
  const int kh = lane >> 4;
  v16bf f;
#pragma unroll
  for (int e = 0; e < 16; ++e) f[e] = b[n * ld + kh * 16 + e];
  return f;
}

// B 32x16 bf16 from an LDS tile stored [k][n]
static __device__ __forceinline__ v16bf frag_b_load_kn(const __bf16* b, int ld) {
  const int lane = threadIdx.x & 31;
  const int n  = lane & 15;
  const int kh = lane >> 4;
  v16bf f;
#pragma unroll
  for (int e = 0; e < 16; ++e) f[e] = b[(kh * 16 + e) * ld + n];
  return f;
}

static __device__ __forceinline__ v8f v8f_zero() {
  v8f z = {0.f, 0.f, 0.f, 0.f, 0.f, 0.f, 0.f, 0.f};
  return z;
}

// (b,h,n,c) scatter offset for qkv epilogue; col is the 0..255 (h,ch) index
static __device__ __forceinline__ size_t qkv_off(int row, int col) {
  const int hI = (col >> 5) & 7;
  const int cc = col & 31;
  const int b  = row >> 12;
  const int n  = row & 4095;
  return ((size_t)(b * 8 + hI) * 4096 + n) * 32 + cc;
}

// ---------------------------------------------------------------------------
// Kernel 1: qkv = x @ w_qkv^T  (M=65536, N=768, K=256)
// blockIdx.y selects 128 output cols; cols [0,256)=q, [256,512)=k, [512,768)=v
// q,k -> bf16 [BH][4096][32], v -> f32 [BH][4096][32]
// ---------------------------------------------------------------------------
__global__ __launch_bounds__(256) void qkv_gemm_kernel(
    const float* __restrict__ x, const float* __restrict__ wqkv,
    __bf16* __restrict__ qb, __bf16* __restrict__ kb, float* __restrict__ vf) {
  __shared__ __bf16 As[128 * 32];
  __shared__ __bf16 Bs[128 * 32];
  const int rowBase = blockIdx.x * 128;
  const int colBase = blockIdx.y * 128;
  const int t    = threadIdx.x;
  const int lane = t & 31;
  const int wave = t >> 5;
  const int wm = wave >> 1;  // 0..3 : 32 rows each
  const int wn = wave & 1;   // 0..1 : 64 cols each

  v8f acc[2][4];
#pragma unroll
  for (int i = 0; i < 2; ++i)
#pragma unroll
    for (int j = 0; j < 4; ++j) acc[i][j] = v8f_zero();

  for (int kbase = 0; kbase < 256; kbase += 32) {
    __syncthreads();
#pragma unroll
    for (int i = 0; i < 4; ++i) {  // A tile: 128x32 f32 -> bf16
      const int idx4 = t + 256 * i;
      const int r  = idx4 >> 3;
      const int kk = (idx4 & 7) * 4;
      const float4 v4 = *(const float4*)(x + (size_t)(rowBase + r) * 256 + kbase + kk);
      __bf16* dst = As + r * 32 + kk;
      dst[0] = (__bf16)v4.x; dst[1] = (__bf16)v4.y;
      dst[2] = (__bf16)v4.z; dst[3] = (__bf16)v4.w;
    }
#pragma unroll
    for (int i = 0; i < 4; ++i) {  // B tile: w_qkv rows [colBase..+128) x 32
      const int idx4 = t + 256 * i;
      const int r  = idx4 >> 3;
      const int kk = (idx4 & 7) * 4;
      const float4 v4 = *(const float4*)(wqkv + (size_t)(colBase + r) * 256 + kbase + kk);
      __bf16* dst = Bs + r * 32 + kk;
      dst[0] = (__bf16)v4.x; dst[1] = (__bf16)v4.y;
      dst[2] = (__bf16)v4.z; dst[3] = (__bf16)v4.w;
    }
    // prefetch next K tile while this one is consumed
    if (kbase + 32 < 256) {
      __builtin_prefetch(x    + (size_t)(rowBase + (t >> 1)) * 256 + kbase + 32 + (t & 1) * 16, 0, 3);
      __builtin_prefetch(wqkv + (size_t)(colBase + (t >> 1)) * 256 + kbase + 32 + (t & 1) * 16, 0, 3);
    }
    __syncthreads();

    v16bf af[2], bq[4];
#pragma unroll
    for (int mt = 0; mt < 2; ++mt)
      af[mt] = frag_a_load(As + (wm * 32 + mt * 16) * 32, 32);
#pragma unroll
    for (int nt = 0; nt < 4; ++nt)
      bq[nt] = frag_b_load_nk(Bs + (wn * 64 + nt * 16) * 32, 32);
#pragma unroll
    for (int mt = 0; mt < 2; ++mt)
#pragma unroll
      for (int nt = 0; nt < 4; ++nt)
        acc[mt][nt] = __builtin_amdgcn_wmma_f32_16x16x32_bf16(
            false, af[mt], false, bq[nt], (short)0, acc[mt][nt], false, false);
  }

  const int kh = lane >> 4;
  const int nc = lane & 15;
  const int n3 = colBase >> 8;  // block-uniform: 0=q, 1=k, 2=v
  if (n3 == 0) {
#pragma unroll
    for (int mt = 0; mt < 2; ++mt)
#pragma unroll
      for (int nt = 0; nt < 4; ++nt)
#pragma unroll
        for (int r = 0; r < 8; ++r) {
          const int row = rowBase + wm * 32 + mt * 16 + kh * 8 + r;
          const int col = colBase + wn * 64 + nt * 16 + nc;
          qb[qkv_off(row, col)] = (__bf16)acc[mt][nt][r];
        }
  } else if (n3 == 1) {
#pragma unroll
    for (int mt = 0; mt < 2; ++mt)
#pragma unroll
      for (int nt = 0; nt < 4; ++nt)
#pragma unroll
        for (int r = 0; r < 8; ++r) {
          const int row = rowBase + wm * 32 + mt * 16 + kh * 8 + r;
          const int col = colBase + wn * 64 + nt * 16 + nc;
          kb[qkv_off(row, col)] = (__bf16)acc[mt][nt][r];
        }
  } else {
#pragma unroll
    for (int mt = 0; mt < 2; ++mt)
#pragma unroll
      for (int nt = 0; nt < 4; ++nt)
#pragma unroll
        for (int r = 0; r < 8; ++r) {
          const int row = rowBase + wm * 32 + mt * 16 + kh * 8 + r;
          const int col = colBase + wn * 64 + nt * 16 + nc;
          vf[qkv_off(row, col)] = acc[mt][nt][r];
        }
  }
}

// ---------------------------------------------------------------------------
// Kernel 2: softmax stats over tokens: m[bh,c] = max_n k, s[bh,c] = sum exp(k-m)
// ---------------------------------------------------------------------------
__global__ __launch_bounds__(256) void kstats_kernel(
    const __bf16* __restrict__ kb, float* __restrict__ mout, float* __restrict__ sout) {
  __shared__ float ms[256], ss[256];
  const int bh = blockIdx.x;
  const int c = threadIdx.x & 31;
  const int g = threadIdx.x >> 5;
  float m = -3.0e38f, s = 0.f;
  for (int n = g; n < 4096; n += 8) {
    const float xv = (float)kb[((size_t)bh * 4096 + n) * 32 + c];
    const float mn = fmaxf(m, xv);
    s = s * __expf(m - mn) + __expf(xv - mn);
    m = mn;
  }
  ms[threadIdx.x] = m;
  ss[threadIdx.x] = s;
  __syncthreads();
  if (g == 0) {
#pragma unroll
    for (int i = 1; i < 8; ++i) {
      const float m2 = ms[i * 32 + c], s2 = ss[i * 32 + c];
      const float mn = fmaxf(m, m2);
      s = s * __expf(m - mn) + s2 * __expf(m2 - mn);
      m = mn;
    }
    mout[bh * 32 + c] = m;
    sout[bh * 32 + c] = s;
  }
}

// ---------------------------------------------------------------------------
// Kernel 3: kv[bh,c,cv] = sum_n exp(k[n,c]-m_c) * v[n,cv]   (1/s folded later)
// ---------------------------------------------------------------------------
__global__ __launch_bounds__(256) void kv_gemm_kernel(
    const __bf16* __restrict__ kb, const float* __restrict__ vf,
    const float* __restrict__ mbuf, float* __restrict__ kv) {
  __shared__ __bf16 a_sh[8][32 * 32];
  __shared__ __bf16 b_sh[8][32 * 32];
  __shared__ float m_lds[32];
  const int bh = blockIdx.x;
  const int t = threadIdx.x;
  const int wave = t >> 5;
  const int lane = t & 31;
  if (t < 32) m_lds[t] = mbuf[bh * 32 + t];
  __syncthreads();

  v8f acc[2][2];
#pragma unroll
  for (int i = 0; i < 2; ++i)
#pragma unroll
    for (int j = 0; j < 2; ++j) acc[i][j] = v8f_zero();

  const size_t base = (size_t)bh * 4096 * 32;
  for (int step = 0; step < 16; ++step) {
    const int n = wave * 512 + step * 32 + lane;
    const __bf16* krow = kb + base + (size_t)n * 32;
    const float*  vrow = vf + base + (size_t)n * 32;
#pragma unroll
    for (int ci = 0; ci < 32; ++ci) {
      a_sh[wave][ci * 32 + lane] = (__bf16)__expf((float)krow[ci] - m_lds[ci]);  // A[c][n]
      b_sh[wave][lane * 32 + ci] = (__bf16)vrow[ci];                             // B[n][cv]
    }
    v16bf af0 = frag_a_load(&a_sh[wave][0], 32);
    v16bf af1 = frag_a_load(&a_sh[wave][16 * 32], 32);
    v16bf bf0 = frag_b_load_kn(&b_sh[wave][0], 32);
    v16bf bf1 = frag_b_load_kn(&b_sh[wave][16], 32);
    acc[0][0] = __builtin_amdgcn_wmma_f32_16x16x32_bf16(false, af0, false, bf0, (short)0, acc[0][0], false, false);
    acc[0][1] = __builtin_amdgcn_wmma_f32_16x16x32_bf16(false, af0, false, bf1, (short)0, acc[0][1], false, false);
    acc[1][0] = __builtin_amdgcn_wmma_f32_16x16x32_bf16(false, af1, false, bf0, (short)0, acc[1][0], false, false);
    acc[1][1] = __builtin_amdgcn_wmma_f32_16x16x32_bf16(false, af1, false, bf1, (short)0, acc[1][1], false, false);
  }

  const int kh = lane >> 4;
  const int nc = lane & 15;
#pragma unroll
  for (int mt = 0; mt < 2; ++mt)
#pragma unroll
    for (int nt = 0; nt < 2; ++nt)
#pragma unroll
      for (int r = 0; r < 8; ++r) {
        const int crow = mt * 16 + kh * 8 + r;
        const int cvcol = nt * 16 + nc;
        atomicAdd(&kv[(size_t)bh * 1024 + crow * 32 + cvcol], acc[mt][nt][r]);
      }
}

// ---------------------------------------------------------------------------
// Kernel 4: y = (scale/s) * q@kv + q * dwconv3x3(v),  y bf16 (B,N,C)
// ---------------------------------------------------------------------------
__global__ __launch_bounds__(256) void fuse_kernel(
    const __bf16* __restrict__ qb, const float* __restrict__ vf,
    const float* __restrict__ kvg, const float* __restrict__ sbuf,
    const float* __restrict__ convw, __bf16* __restrict__ yb) {
  __shared__ __bf16 vt[10 * 64 * 32];  // rows-with-halo x width x ch
  __shared__ float kvs[32 * 32];
  __shared__ float wco[288];
  const int bh = blockIdx.x;
  const int rb = blockIdx.y;  // row-block 0..7
  const int t = threadIdx.x;
  const int hI = bh & 7, b = bh >> 3;

  for (int idx = t; idx < 1024; idx += 256) {
    const int c = idx >> 5;
    kvs[idx] = kvg[(size_t)bh * 1024 + idx] * (ATT_SCALE / sbuf[bh * 32 + c]);
  }
  for (int idx = t; idx < 288; idx += 256) wco[idx] = convw[hI * 288 + idx];

  const size_t vbase = (size_t)bh * 131072;
  for (int idx = t; idx < 20480; idx += 256) {
    const int i = idx >> 11;       // halo row 0..9
    const int rem = idx & 2047;    // (x, c)
    const int rimg = rb * 8 - 1 + i;
    const float val = (rimg >= 0 && rimg < 64) ? vf[vbase + (size_t)rimg * 2048 + rem] : 0.f;
    vt[idx] = (__bf16)val;
  }
  __syncthreads();

  const int cv = t & 31;
  const int w  = t >> 5;  // image row within block
  float wreg[9];
#pragma unroll
  for (int j = 0; j < 9; ++j) wreg[j] = wco[cv * 9 + j];
  float kcol[32];
#pragma unroll
  for (int c = 0; c < 32; ++c) kcol[c] = kvs[c * 32 + cv];

  const int yy = rb * 8 + w;
  const size_t qbase = (size_t)bh * 131072;
  for (int xx = 0; xx < 64; ++xx) {
    const int n = yy * 64 + xx;
    const float qv = (float)qb[qbase + (size_t)n * 32 + cv];
    float cacc = 0.f;
#pragma unroll
    for (int ky = 0; ky < 3; ++ky) {
      const __bf16* rowp = &vt[((w + ky) * 64) * 32 + cv];
#pragma unroll
      for (int kx = 0; kx < 3; ++kx) {
        const int xxx = xx + kx - 1;
        if (xxx >= 0 && xxx < 64)
          cacc += wreg[ky * 3 + kx] * (float)rowp[xxx * 32];
      }
    }
    float fac = 0.f;
#pragma unroll
    for (int c = 0; c < 32; ++c)
      fac += __shfl(qv, c, 32) * kcol[c];
    const float yv = fac + qv * cacc;
    yb[((size_t)(b * 4096 + n)) * 256 + hI * 32 + cv] = (__bf16)yv;
  }
}

// ---------------------------------------------------------------------------
// Kernel 5: out = y @ w_proj^T + b_proj  (M=65536, N=256, K=256), f32 out
// A tile is a pure bf16 copy -> use CDNA5 async global->LDS DMA when available
// ---------------------------------------------------------------------------
__global__ __launch_bounds__(256) void proj_gemm_kernel(
    const __bf16* __restrict__ yb, const float* __restrict__ wproj,
    const float* __restrict__ bproj, float* __restrict__ out) {
  __shared__ __bf16 As[128 * 32];
  __shared__ __bf16 Bs[128 * 32];
  const int rowBase = blockIdx.x * 128;
  const int colBase = blockIdx.y * 128;
  const int t    = threadIdx.x;
  const int lane = t & 31;
  const int wave = t >> 5;
  const int wm = wave >> 1;
  const int wn = wave & 1;

  v8f acc[2][4];
#pragma unroll
  for (int i = 0; i < 2; ++i)
#pragma unroll
    for (int j = 0; j < 4; ++j) acc[i][j] = v8f_zero();

  for (int kbase = 0; kbase < 256; kbase += 32) {
    __syncthreads();
    // A tile: y bf16 128x32 -- straight copy
#if HAVE_ASYNC_LDS
#pragma unroll
    for (int i = 0; i < 2; ++i) {
      const int idx8 = t + 256 * i;
      const int r  = idx8 >> 2;
      const int kk = (idx8 & 3) * 8;
      async_copy_b128(yb + (size_t)(rowBase + r) * 256 + kbase + kk,
                      As + r * 32 + kk);
    }
#else
#pragma unroll
    for (int i = 0; i < 2; ++i) {
      const int idx8 = t + 256 * i;
      const int r  = idx8 >> 2;
      const int kk = (idx8 & 3) * 8;
      *(v8bf*)(As + r * 32 + kk) =
          *(const v8bf*)(yb + (size_t)(rowBase + r) * 256 + kbase + kk);
    }
#endif
#pragma unroll
    for (int i = 0; i < 4; ++i) {  // B tile: w_proj f32 -> bf16
      const int idx4 = t + 256 * i;
      const int r  = idx4 >> 3;
      const int kk = (idx4 & 7) * 4;
      const float4 v4 = *(const float4*)(wproj + (size_t)(colBase + r) * 256 + kbase + kk);
      __bf16* dst = Bs + r * 32 + kk;
      dst[0] = (__bf16)v4.x; dst[1] = (__bf16)v4.y;
      dst[2] = (__bf16)v4.z; dst[3] = (__bf16)v4.w;
    }
    if (kbase + 32 < 256) {
      __builtin_prefetch(yb    + (size_t)(rowBase + (t >> 1)) * 256 + kbase + 32 + (t & 1) * 16, 0, 3);
      __builtin_prefetch(wproj + (size_t)(colBase + (t >> 1)) * 256 + kbase + 32 + (t & 1) * 16, 0, 3);
    }
#if HAVE_ASYNC_LDS
    __builtin_amdgcn_s_wait_asynccnt(0);
#endif
    __syncthreads();

    v16bf af[2], bq[4];
#pragma unroll
    for (int mt = 0; mt < 2; ++mt)
      af[mt] = frag_a_load(As + (wm * 32 + mt * 16) * 32, 32);
#pragma unroll
    for (int nt = 0; nt < 4; ++nt)
      bq[nt] = frag_b_load_nk(Bs + (wn * 64 + nt * 16) * 32, 32);
#pragma unroll
    for (int mt = 0; mt < 2; ++mt)
#pragma unroll
      for (int nt = 0; nt < 4; ++nt)
        acc[mt][nt] = __builtin_amdgcn_wmma_f32_16x16x32_bf16(
            false, af[mt], false, bq[nt], (short)0, acc[mt][nt], false, false);
  }

  const int kh = lane >> 4;
  const int nc = lane & 15;
#pragma unroll
  for (int mt = 0; mt < 2; ++mt)
#pragma unroll
    for (int nt = 0; nt < 4; ++nt)
#pragma unroll
      for (int r = 0; r < 8; ++r) {
        const int row = rowBase + wm * 32 + mt * 16 + kh * 8 + r;
        const int col = colBase + wn * 64 + nt * 16 + nc;
        out[(size_t)row * 256 + col] = acc[mt][nt][r] + bproj[col];
      }
}

// ---------------------------------------------------------------------------
// Host-side launcher
// ---------------------------------------------------------------------------
extern "C" void kernel_launch(void* const* d_in, const int* in_sizes, int n_in,
                              void* d_out, int out_size, void* d_ws, size_t ws_size,
                              hipStream_t stream) {
  (void)in_sizes; (void)n_in; (void)out_size; (void)ws_size;
  const float* x     = (const float*)d_in[0];
  const float* wqkv  = (const float*)d_in[1];
  const float* wproj = (const float*)d_in[2];
  const float* bproj = (const float*)d_in[3];
  const float* convw = (const float*)d_in[4];
  float* out = (float*)d_out;

  char* ws = (char*)d_ws;
  // workspace layout (bytes)
  __bf16* qb = (__bf16*)(ws + 0);            //  32 MB  q  bf16 [128][4096][32]
  __bf16* kb = (__bf16*)(ws + 33554432ull);  //  32 MB  k  bf16
  float*  vf = (float*)(ws + 67108864ull);   //  64 MB  v  f32
  float*  mb = (float*)(ws + 134217728ull);  //  16 KB  max
  float*  sb = (float*)(ws + 134234112ull);  //  16 KB  sumexp
  float*  kv = (float*)(ws + 134250496ull);  // 512 KB  kv  [128][32][32]
  __bf16* yb = (__bf16*)(ws + 135266304ull); //  32 MB  y  bf16 [65536][256]

  (void)hipMemsetAsync(kv, 0, 128 * 32 * 32 * sizeof(float), stream);
  qkv_gemm_kernel<<<dim3(512, 6), 256, 0, stream>>>(x, wqkv, qb, kb, vf);
  kstats_kernel<<<dim3(128), 256, 0, stream>>>(kb, mb, sb);
  kv_gemm_kernel<<<dim3(128), 256, 0, stream>>>(kb, vf, mb, kv);
  fuse_kernel<<<dim3(128, 8), 256, 0, stream>>>(qb, vf, kv, sb, convw, yb);
  proj_gemm_kernel<<<dim3(512, 2), 256, 0, stream>>>(yb, wproj, bproj, out);
}
